// Qwen3VL2BCrossAttentionKernel_59622736003267
// MI455X (gfx1250) — matrix operand order repaired and checked by reading, not compiled, not run on hardware
//
#include <hip/hip_runtime.h>
#include <hip/hip_bf16.h>
#include <math.h>

typedef __bf16 bf16;
typedef __attribute__((ext_vector_type(16))) __bf16 v16bf;
typedef __attribute__((ext_vector_type(8)))  __bf16 v8bf;
typedef __attribute__((ext_vector_type(8)))  float  v8f;
typedef unsigned int u32;
typedef __attribute__((ext_vector_type(4))) unsigned int u32x4;
typedef __attribute__((ext_vector_type(8))) int          i32x8;
typedef __attribute__((ext_vector_type(4))) int          i32x4;

#define D_MODEL   2048
#define N_HEADS   16
#define HEAD_DIM  128
#define KV_LEN    3072
#define ATTN_SCALE 0.08838834764831845f   // 128^-0.5
#define LN_EPS    1e-5f

#if __has_builtin(__builtin_amdgcn_tensor_load_to_lds)
#define HAVE_TDM 1
#else
#define HAVE_TDM 0
#endif

union BF16Frag { v16bf v; v8bf h[2]; };

__device__ __forceinline__ v8bf ld8(const bf16* p) { return *(const v8bf*)p; }

__device__ __forceinline__ v8f wmma_bf16(v16bf a, v16bf b, v8f c) {
  // (neg_a, A, neg_b, B, c_mod, C, reuse_a, reuse_b)
  return __builtin_amdgcn_wmma_f32_16x16x32_bf16(false, a, false, b, (short)0, c, false, false);
}

__device__ __forceinline__ void wait_tensorcnt0() {
#if __has_builtin(__builtin_amdgcn_s_wait_tensorcnt)
  __builtin_amdgcn_s_wait_tensorcnt(0);
#else
  asm volatile("s_wait_tensorcnt 0x0" ::: "memory");
#endif
}

#if HAVE_TDM
// ---------------------------------------------------------------------------
// Issue one TDM 2D tile load (TENSOR_LOAD_TO_LDS). D# per CDNA5 ISA §8:
// group0: count=1 | lds_addr | global_addr[56:0] | type=2
// group1: data_size=2B, optional LDS padding (pad_interval 32 DW, +4 DW ->
//         reproduces a 72-element bf16 row stride for 64-element rows),
//         tensor_dim0/1, tile_dim0/1 (tile_dim2=0 => 2D), dim0 stride.
// Uniform (blockIdx-derived) operands -> backend materializes SGPR groups.
// This toolchain exposes the 6-arg builtin:
//   (u32x4 g0, i32x8 g1, i32x4 g2, i32x4 g3, i32x8 extra, int cpol)
// groups 2/3 (+extra) are zero for a 2D tile.
// ---------------------------------------------------------------------------
__device__ __forceinline__ void tdm_load_2d(const void* gaddr, u32 lds_byte_off,
                                            u32 tensor_d0, u32 tensor_d1,
                                            u32 tile_d0, u32 tile_d1,
                                            u32 stride_d0, bool pad) {
  const unsigned long long ga = (unsigned long long)gaddr;
  u32x4 g0;
  g0[0] = 1u;                                            // count=1 (user D#)
  g0[1] = lds_byte_off;                                  // lds_addr [63:32]
  g0[2] = (u32)ga;                                       // global_addr lo
  g0[3] = (u32)((ga >> 32) & 0x1FFFFFFu) | (2u << 30);   // addr[56:32] | type=2
  u32 d0 = (1u << 16);                                   // data_size=1 -> 2 bytes
  if (pad) d0 |= (1u << 20) | (4u << 22) | (3u << 25);   // pad_en, every 32 DW, +4 DW
  i32x8 g1;
  g1[0] = (int)d0;
  g1[1] = (int)((tensor_d0 & 0xFFFFu) << 16);                                  // dim0 lo16
  g1[2] = (int)(((tensor_d0 >> 16) & 0xFFFFu) | ((tensor_d1 & 0xFFFFu) << 16)); // dim0 hi | dim1 lo
  g1[3] = (int)(((tensor_d1 >> 16) & 0xFFFFu) | (tile_d0 << 16));               // dim1 hi | tile_dim0
  g1[4] = (int)(tile_d1 & 0xFFFFu);                                            // tile_dim1, tile_dim2=0
  g1[5] = (int)stride_d0;                                                      // dim0_stride lo32
  g1[6] = 0;
  g1[7] = 0;
  const i32x4 z4 = {0, 0, 0, 0};                         // 2D: groups 2/3 unused
  const i32x8 z8 = {0, 0, 0, 0, 0, 0, 0, 0};
  __builtin_amdgcn_tensor_load_to_lds(g0, g1, z4, z4, z8, 0);
}
#endif

// ---------------------------------------------------------------------------
// LayerNorm f32 -> bf16, with output-row remap so text/image key rows land
// interleaved per batch ([b, 0..2047]=text, [b, 2048..3071]=image).
// ---------------------------------------------------------------------------
__global__ __launch_bounds__(256)
void ln_bf16_kernel(const float* __restrict__ x, const float* __restrict__ g,
                    const float* __restrict__ beta, bf16* __restrict__ y,
                    int s_in, int out_batch_stride, int out_offset) {
  const int row = blockIdx.x;
  const int bb  = row / s_in;
  const int s   = row - bb * s_in;
  const float* xr = x + (size_t)row * D_MODEL;
  bf16* yr = y + ((size_t)bb * out_batch_stride + out_offset + s) * D_MODEL;

  float v[8];
  float sum = 0.f, sq = 0.f;
  const int base = threadIdx.x * 8;
#pragma unroll
  for (int i = 0; i < 8; ++i) { v[i] = xr[base + i]; sum += v[i]; sq += v[i] * v[i]; }

  __shared__ float sred[2][8];
  const int lane = threadIdx.x & 31, wid = threadIdx.x >> 5;
#pragma unroll
  for (int off = 16; off > 0; off >>= 1) {
    sum += __shfl_down(sum, off, 32);
    sq  += __shfl_down(sq,  off, 32);
  }
  if (lane == 0) { sred[0][wid] = sum; sred[1][wid] = sq; }
  __syncthreads();
  if (threadIdx.x == 0) {
    float a = 0.f, c = 0.f;
    for (int i = 0; i < 8; ++i) { a += sred[0][i]; c += sred[1][i]; }
    sred[0][0] = a; sred[1][0] = c;
  }
  __syncthreads();
  const float mu   = sred[0][0] * (1.f / D_MODEL);
  const float var  = sred[1][0] * (1.f / D_MODEL) - mu * mu;
  const float rstd = rsqrtf(var + LN_EPS);
#pragma unroll
  for (int i = 0; i < 8; ++i) {
    const int c = base + i;
    yr[c] = (bf16)((v[i] - mu) * rstd * g[c] + beta[c]);
  }
}

// ---------------------------------------------------------------------------
// Weight convert f32 -> bf16 (n divisible by 2048)
// ---------------------------------------------------------------------------
__global__ __launch_bounds__(256)
void f32_to_bf16_kernel(const float* __restrict__ x, bf16* __restrict__ y, int n) {
  const int i = (blockIdx.x * 256 + threadIdx.x) * 8;
  if (i + 8 <= n) {
#pragma unroll
    for (int j = 0; j < 8; ++j) y[i + j] = (bf16)x[i + j];
  }
}

// ---------------------------------------------------------------------------
// NT GEMM: Y[M,2048] = A[M,2048] @ W[2048,2048]^T, bf16 in, f32 accumulate.
// Block = 256 threads (8 waves), tile 128x128, BK=64, double-buffered LDS.
// TDM path: wave 0 DMAs the A tile, wave 1 the B tile (TENSOR_LOAD_TO_LDS
// with hardware LDS padding), each waits s_wait_tensorcnt 0, block barrier
// publishes the tile; stage i+1's DMA overlaps stage i's WMMA phase.
// Fallback: register-pipelined global_load_b128 + ds_store_b128.
// MODE 0: bf16 row-major out, scaled.  MODE 1: bf16 transposed-per-head out
// (VT[b,h,d,tok]).  MODE 2: f32 row-major out.
// ---------------------------------------------------------------------------
template <int MODE>
__global__ __launch_bounds__(256)
void gemm_nt_bf16(const bf16* __restrict__ A, const bf16* __restrict__ Bw,
                  bf16* __restrict__ outB, float* __restrict__ outF,
                  int M, float scale) {
  __shared__ __align__(16) bf16 As[2][128][72];
  __shared__ __align__(16) bf16 Bs[2][128][72];

  const int tid  = threadIdx.x;
  const int lane = tid & 31, w = tid >> 5;
  const int half = lane >> 4, ln16 = lane & 15;
  const int wm = w & 3, wn = w >> 2;
  const int blockM = blockIdx.y * 128, blockN = blockIdx.x * 128;

  v8f acc[2][4];
  const v8f zacc = {};
#pragma unroll
  for (int mt = 0; mt < 2; ++mt)
#pragma unroll
    for (int nt = 0; nt < 4; ++nt) acc[mt][nt] = zacc;

#if HAVE_TDM
  auto issueStage = [&](int k0, int buf) {
    if (w == 0)
      tdm_load_2d(&A[(size_t)blockM * D_MODEL + k0], (u32)(uintptr_t)&As[buf][0][0],
                  D_MODEL, (u32)M, 64, 128, D_MODEL, true);
    else if (w == 1)
      tdm_load_2d(&Bw[(size_t)blockN * D_MODEL + k0], (u32)(uintptr_t)&Bs[buf][0][0],
                  D_MODEL, D_MODEL, 64, 128, D_MODEL, true);
  };
  issueStage(0, 0);
#else
  v8bf aReg[4], bReg[4];
  auto loadStage = [&](int k0) {
#pragma unroll
    for (int j = 0; j < 4; ++j) {
      const int c = tid + 256 * j;
      const int r = c >> 3, cc = (c & 7) * 8;
      const size_t ao = (size_t)(blockM + r) * D_MODEL + k0 + cc;
      const size_t bo = (size_t)(blockN + r) * D_MODEL + k0 + cc;
      aReg[j] = ld8(&A[ao]);
      bReg[j] = ld8(&Bw[bo]);
      __builtin_prefetch(&A[ao]  + 128, 0, 3);
      __builtin_prefetch(&Bw[bo] + 128, 0, 3);
    }
  };
  auto storeStage = [&](int buf) {
#pragma unroll
    for (int j = 0; j < 4; ++j) {
      const int c = tid + 256 * j;
      const int r = c >> 3, cc = (c & 7) * 8;
      *(v8bf*)&As[buf][r][cc] = aReg[j];
      *(v8bf*)&Bs[buf][r][cc] = bReg[j];
    }
  };
  loadStage(0);
#endif

  int ib = 0;
  for (int k0 = 0; k0 < D_MODEL; k0 += 64, ib ^= 1) {
#if HAVE_TDM
    if (w < 2) wait_tensorcnt0();          // my stage-ib DMA has landed in LDS
    __syncthreads();                       // publish to all waves
    if (k0 + 64 < D_MODEL) issueStage(k0 + 64, ib ^ 1);  // overlap with WMMA
#else
    storeStage(ib);
    __syncthreads();
    if (k0 + 64 < D_MODEL) loadStage(k0 + 64);
#endif

#pragma unroll
    for (int kk = 0; kk < 64; kk += 32) {
      BF16Frag af[2], bf[4];
#pragma unroll
      for (int mt = 0; mt < 2; ++mt) {
        const bf16* p = &As[ib][wm * 32 + mt * 16 + ln16][kk + 8 * half];
        af[mt].h[0] = ld8(p); af[mt].h[1] = ld8(p + 16);
      }
#pragma unroll
      for (int nt = 0; nt < 4; ++nt) {
        const bf16* p = &Bs[ib][wn * 64 + nt * 16 + ln16][kk + 8 * half];
        bf[nt].h[0] = ld8(p); bf[nt].h[1] = ld8(p + 16);
      }
#pragma unroll
      for (int mt = 0; mt < 2; ++mt)
#pragma unroll
        for (int nt = 0; nt < 4; ++nt)
          acc[mt][nt] = wmma_bf16(af[mt].v, bf[nt].v, acc[mt][nt]);
    }
    // no trailing barrier: double buffering + the top-of-loop barrier
    // guarantee the buffer being refilled was last read one stage ago.
  }

#pragma unroll
  for (int mt = 0; mt < 2; ++mt)
#pragma unroll
    for (int nt = 0; nt < 4; ++nt)
#pragma unroll
      for (int r = 0; r < 8; ++r) {
        const int m = blockM + wm * 32 + mt * 16 + r + 8 * half;
        const int n = blockN + wn * 64 + nt * 16 + ln16;
        const float val = acc[mt][nt][r];
        if constexpr (MODE == 0) {
          outB[(size_t)m * D_MODEL + n] = (bf16)(val * scale);
        } else if constexpr (MODE == 1) {
          const int b = m / KV_LEN, tok = m - b * KV_LEN;   // constant division
          const int h = n >> 7, d = n & 127;
          outB[(((size_t)(b * N_HEADS + h) * HEAD_DIM) + d) * (size_t)KV_LEN + tok] = (bf16)val;
        } else {
          outF[(size_t)m * D_MODEL + n] = val;
        }
      }
}

// ---------------------------------------------------------------------------
// Flash attention. Q pre-scaled by 128^-0.5. Block = 8 waves sharing one
// (b,h); each wave owns a 16-row q tile. Per 64-token kv chunk the block
// cooperatively stages K ([tok][d]) and V ([d][tok], from VT) into LDS once,
// with next-chunk global loads issued into registers during the WMMA phase.
// Softmax stats per lane-half (rows = vreg + 8*(lane>>4)); butterflies with
// masks<=8 stay inside each 16-lane half. P goes through per-wave LDS to
// convert C/D layout to A-fragment layout (DS ops of a wave are in-order;
// s_wait_dscnt 0 before the re-read).
// ---------------------------------------------------------------------------
__global__ __launch_bounds__(256)
void attn_kernel(const bf16* __restrict__ Q, const bf16* __restrict__ K,
                 const bf16* __restrict__ VT, bf16* __restrict__ O,
                 int qRowBase, int Sq) {
  __shared__ __align__(16) bf16 Kt[64][136];    // [tok][d]   17408 B
  __shared__ __align__(16) bf16 Vt[128][72];    // [d][tok]   18432 B
  __shared__ __align__(16) bf16 Ps[8][16][72];  // per-wave P 18432 B
  const int tid  = threadIdx.x;
  const int lane = tid & 31, w = tid >> 5;
  const int half = lane >> 4, ln16 = lane & 15;
  const int b = blockIdx.z, h = blockIdx.y;
  const int headCol = h * HEAD_DIM;
  const int qRow0 = qRowBase + b * Sq + blockIdx.x * 128 + w * 16;

  const bf16* Kbase = K  + (size_t)(b * KV_LEN) * D_MODEL + headCol;
  const bf16* Vbase = VT + (size_t)((b * N_HEADS + h) * HEAD_DIM) * KV_LEN;

  // resident Q fragments: 4 K-chunks of 32 covering the 128-dim head
  BF16Frag qf[4];
  const bf16* qrow = Q + (size_t)(qRow0 + ln16) * D_MODEL + headCol + 8 * half;
#pragma unroll
  for (int dc = 0; dc < 4; ++dc) {
    qf[dc].h[0] = ld8(qrow + dc * 32);
    qf[dc].h[1] = ld8(qrow + dc * 32 + 16);
  }

  v8f o[8];
  const v8f zacc = {};
  float rm[8], rl[8];
#pragma unroll
  for (int i = 0; i < 8; ++i) { o[i] = zacc; rm[i] = -1e30f; rl[i] = 0.f; }

  // cooperative staging: 256 threads move 64x128 K elems + 128x64 V elems
  v8bf kreg[4], vreg[4];
  auto loadStage = [&](int kb) {
#pragma unroll
    for (int j = 0; j < 4; ++j) {
      const int c = tid + 256 * j;
      kreg[j] = ld8(&Kbase[(size_t)(kb + (c >> 4)) * D_MODEL + (c & 15) * 8]);
      vreg[j] = ld8(&Vbase[(size_t)(c >> 3) * KV_LEN + kb + (c & 7) * 8]);
    }
  };
  auto storeStage = [&]() {
#pragma unroll
    for (int j = 0; j < 4; ++j) {
      const int c = tid + 256 * j;
      *(v8bf*)&Kt[c >> 4][(c & 15) * 8] = kreg[j];
      *(v8bf*)&Vt[c >> 3][(c & 7) * 8]  = vreg[j];
    }
  };

  loadStage(0);
  for (int kb = 0; kb < KV_LEN; kb += 64) {
    storeStage();
    __syncthreads();
    if (kb + 64 < KV_LEN) loadStage(kb + 64);    // overlap with compute

    // S tiles: 4 n-tiles of 16 kv tokens, K fragments from LDS
    v8f s[4];
#pragma unroll
    for (int j = 0; j < 4; ++j) {
      const bf16* krow = &Kt[j * 16 + ln16][8 * half];
      v8f a = zacc;
#pragma unroll
      for (int dc = 0; dc < 4; ++dc) {
        BF16Frag kf;
        kf.h[0] = ld8(krow + dc * 32);
        kf.h[1] = ld8(krow + dc * 32 + 16);
        a = wmma_bf16(qf[dc].v, kf.v, a);
      }
      s[j] = a;
    }

    // running softmax
    float nm[8], rs[8], corr[8];
#pragma unroll
    for (int v = 0; v < 8; ++v)
      nm[v] = fmaxf(fmaxf(s[0][v], s[1][v]), fmaxf(s[2][v], s[3][v]));
#pragma unroll
    for (int off = 8; off > 0; off >>= 1)
#pragma unroll
      for (int v = 0; v < 8; ++v) nm[v] = fmaxf(nm[v], __shfl_xor(nm[v], off, 32));
#pragma unroll
    for (int v = 0; v < 8; ++v) {
      nm[v]   = fmaxf(nm[v], rm[v]);
      corr[v] = __expf(rm[v] - nm[v]);
      rm[v]   = nm[v];
      rs[v]   = 0.f;
    }
#pragma unroll
    for (int j = 0; j < 4; ++j)
#pragma unroll
      for (int v = 0; v < 8; ++v) {
        const float p = __expf(s[j][v] - nm[v]);
        s[j][v] = p;
        rs[v] += p;
      }
#pragma unroll
    for (int off = 8; off > 0; off >>= 1)
#pragma unroll
      for (int v = 0; v < 8; ++v) rs[v] += __shfl_xor(rs[v], off, 32);
#pragma unroll
    for (int v = 0; v < 8; ++v) rl[v] = rl[v] * corr[v] + rs[v];
#pragma unroll
    for (int i = 0; i < 8; ++i)
#pragma unroll
      for (int v = 0; v < 8; ++v) o[i][v] *= corr[v];

    // C/D layout -> A layout via per-wave LDS round trip (DS ops in-order)
#pragma unroll
    for (int j = 0; j < 4; ++j)
#pragma unroll
      for (int v = 0; v < 8; ++v)
        Ps[w][v + 8 * half][j * 16 + ln16] = (bf16)s[j][v];
    asm volatile("s_wait_dscnt 0x0" ::: "memory");

    BF16Frag pf[2];
#pragma unroll
    for (int kc = 0; kc < 2; ++kc) {
      const bf16* pp = &Ps[w][ln16][kc * 32 + 8 * half];
      pf[kc].h[0] = ld8(pp);
      pf[kc].h[1] = ld8(pp + 16);
    }

    // O += P @ V, V fragments from LDS ([d][tok], contiguous along tokens)
#pragma unroll
    for (int i = 0; i < 8; ++i)
#pragma unroll
      for (int kc = 0; kc < 2; ++kc) {
        const bf16* vrow = &Vt[i * 16 + ln16][kc * 32 + 8 * half];
        BF16Frag vf;
        vf.h[0] = ld8(vrow);
        vf.h[1] = ld8(vrow + 16);
        o[i] = wmma_bf16(pf[kc].v, vf.v, o[i]);
      }
    __syncthreads();   // protect Kt/Vt before next chunk's storeStage
  }

#pragma unroll
  for (int v = 0; v < 8; ++v) rl[v] = 1.f / rl[v];
#pragma unroll
  for (int i = 0; i < 8; ++i)
#pragma unroll
    for (int v = 0; v < 8; ++v) {
      const int row = qRow0 + v + 8 * half;
      const int col = headCol + i * 16 + ln16;
      O[(size_t)row * D_MODEL + col] = (bf16)(o[i][v] * rl[v]);
    }
}

// ---------------------------------------------------------------------------
extern "C" void kernel_launch(void* const* d_in, const int* in_sizes, int n_in,
                              void* d_out, int out_size, void* d_ws, size_t ws_size,
                              hipStream_t stream) {
  (void)in_sizes; (void)n_in; (void)out_size; (void)ws_size;
  const float* q_text  = (const float*)d_in[0];
  const float* q_image = (const float*)d_in[1];
  const float* k_text  = (const float*)d_in[2];
  const float* k_image = (const float*)d_in[3];
  const float* Wq = (const float*)d_in[4];
  const float* Wk = (const float*)d_in[5];
  const float* Wv = (const float*)d_in[6];
  const float* Wo = (const float*)d_in[7];
  const float* g_t = (const float*)d_in[8];
  const float* b_t = (const float*)d_in[9];
  const float* g_i = (const float*)d_in[10];
  const float* b_i = (const float*)d_in[11];
  float* out = (float*)d_out;

  char* ws = (char*)d_ws;
  size_t off = 0;
  auto alloc = [&](size_t bytes) -> char* {
    char* p = ws + off;
    off += (bytes + 255) & ~(size_t)255;
    return p;
  };
  const size_t TOK = 6144;                        // 2*2048 text + 2*1024 image rows
  bf16* lnQ  = (bf16*)alloc(TOK * D_MODEL * 2);   // reused as attention output O
  bf16* lnK  = (bf16*)alloc(TOK * D_MODEL * 2);   // [b, 0..2047 text | 2048..3071 image]
  bf16* Wq16 = (bf16*)alloc((size_t)D_MODEL * D_MODEL * 2);
  bf16* Wk16 = (bf16*)alloc((size_t)D_MODEL * D_MODEL * 2);
  bf16* Wv16 = (bf16*)alloc((size_t)D_MODEL * D_MODEL * 2);
  bf16* Wo16 = (bf16*)alloc((size_t)D_MODEL * D_MODEL * 2);
  bf16* Qs   = (bf16*)alloc(TOK * D_MODEL * 2);
  bf16* Ks   = (bf16*)alloc(TOK * D_MODEL * 2);
  bf16* VTb  = (bf16*)alloc(TOK * D_MODEL * 2);   // [b,h,d,tok]

  // LayerNorms -> bf16 (with row remap for K concat)
  ln_bf16_kernel<<<4096, 256, 0, stream>>>(q_text,  g_t, b_t, lnQ, 2048, 2048, 0);
  ln_bf16_kernel<<<2048, 256, 0, stream>>>(q_image, g_i, b_i, lnQ, 1024, 1024, 4096);
  ln_bf16_kernel<<<4096, 256, 0, stream>>>(k_text,  g_t, b_t, lnK, 2048, 3072, 0);
  ln_bf16_kernel<<<2048, 256, 0, stream>>>(k_image, g_i, b_i, lnK, 1024, 3072, 2048);

  // Weights -> bf16
  const int WN = D_MODEL * D_MODEL;
  f32_to_bf16_kernel<<<WN / 2048, 256, 0, stream>>>(Wq, Wq16, WN);
  f32_to_bf16_kernel<<<WN / 2048, 256, 0, stream>>>(Wk, Wk16, WN);
  f32_to_bf16_kernel<<<WN / 2048, 256, 0, stream>>>(Wv, Wv16, WN);
  f32_to_bf16_kernel<<<WN / 2048, 256, 0, stream>>>(Wo, Wo16, WN);

  // Projections (WMMA GEMMs), M = 6144
  dim3 gblk(16, 48);  // N/128 x M/128
  gemm_nt_bf16<0><<<gblk, 256, 0, stream>>>(lnQ, Wq16, Qs,  nullptr, 6144, ATTN_SCALE);
  gemm_nt_bf16<0><<<gblk, 256, 0, stream>>>(lnK, Wk16, Ks,  nullptr, 6144, 1.0f);
  gemm_nt_bf16<1><<<gblk, 256, 0, stream>>>(lnK, Wv16, VTb, nullptr, 6144, 1.0f);

  // Flash attention (O aliases lnQ, which is dead after the Q projection)
  bf16* O = lnQ;
  attn_kernel<<<dim3(16, 16, 2), 256, 0, stream>>>(Qs, Ks, VTb, O, 0,    2048);
  attn_kernel<<<dim3(8,  16, 2), 256, 0, stream>>>(Qs, Ks, VTb, O, 4096, 1024);

  // Output projection straight into d_out (rows 0..4095 = text, 4096..6143 = image)
  gemm_nt_bf16<2><<<gblk, 256, 0, stream>>>(O, Wo16, nullptr, out, 6144, 0.f);
}
